// LocalSubgraphTransformer_58737972740402
// MI455X (gfx1250) — matrix operand order repaired and checked by reading, not compiled, not run on hardware
//
#include <hip/hip_runtime.h>
#include <hip/hip_bf16.h>

// ---------------------------------------------------------------------------
// Fused LocalSubgraphTransformer for gfx1250 (MI455X).
// TWO subgraphs per workgroup (256 threads = 8 waves, wave32); all
// activations + edge bias in LDS (~295KB of the 320KB/WGP). GEMMs use
// v_wmma_f32_16x16x32_bf16. Weights are converted fp32->bf16 and transposed
// to [K][N] once (workspace) so B fragments are contiguous 32B loads.
// Each wave owns a 16-wide N strip; every B-fragment load feeds FOUR WMMAs
// (2 subgraphs x 2 M-halves), quartering weight (L2) traffic vs. the naive
// per-tile scheme — the dominant cost at 4MB of bf16 weights re-read per
// block (total ~4GB of L2 reads for the grid vs a ~70MB HBM floor).
// ---------------------------------------------------------------------------

typedef __attribute__((ext_vector_type(16))) __bf16 v16bf;
typedef __attribute__((ext_vector_type(8)))  __bf16 v8bf;
typedef __attribute__((ext_vector_type(8)))  float  v8f;

#define S_TOT   2048
#define K_TOK   32
#define H_DIM   256
#define N_HEAD  8
#define D_HEAD  32
#define N_LAYER 4
#define FFN_DIM 512
#define EPS_E   64
#define E_TOT   (S_TOT * EPS_E)
#define SG_PER_BLK 2

// ---- per-subgraph LDS region (bytes) ---------------------------------------
#define O_X     0        // f32  [32][256] residual            32768
#define O_BIAS  32768    // bf16 [8][32][32] edge bias         16384
#define O_XN    49152    // bf16 [32][256] xn | obf (timeshare)16384
#define O_Q     65536    // bf16 [8][32][32]                   16384
#define O_KT    81920    // bf16 [8][32][32] (d-major)         16384
#define O_V     98304    // bf16 [8][32][32]                   16384
#define O_SCR   114688   // 32KB: f32 bias-accum | bf16 scores(16K)+attn(16K)
                         //       | bf16 ffn-hid [32][512] | f32 final-norm
#define SG_STRIDE 147456
#define OFF_NEG (SG_PER_BLK * SG_STRIDE)   // f32 [64] key-padding mask
#define SMEM_BYTES (SG_PER_BLK * SG_STRIDE + 256)

// ---- A fragment: 16-bit A layout (lane m%16; K chunks at koff, koff+16) ----
__device__ __forceinline__ v16bf load_a_frag(const __bf16* base, int lda,
                                             int mrow, int k0, int koff) {
  const __bf16* pa = base + mrow * lda + k0 + koff;
  v8bf lo = *(const v8bf*)(pa);
  v8bf hi = *(const v8bf*)(pa + 16);
  return __builtin_shufflevector(lo, hi,
      0, 1, 2, 3, 4, 5, 6, 7, 8, 9, 10, 11, 12, 13, 14, 15);
}

// ---- 4 stacked 16x16 tiles (2 subgraphs x 2 M-halves) per B-fragment load --
__device__ __forceinline__ void wmma_tile4_bf16(const __bf16* A0, const __bf16* A1,
                                                int lda, const __bf16* B, int ldb,
                                                int n0, int Kdim, int lane,
                                                v8f& c00, v8f& c01,
                                                v8f& c10, v8f& c11) {
  const int mrow = lane & 15;
  const int koff = (lane >> 4) << 3;
  for (int k0 = 0; k0 < Kdim; k0 += 32) {
    const __bf16* pb = B + (long)(k0 + koff) * ldb + n0 + mrow;  // lane = col N, K in table order
    v16bf b;
    for (int i = 0; i < 8; ++i) { b[i] = pb[(long)i * ldb]; b[i + 8] = pb[(long)(i + 16) * ldb]; }
    v16bf a;
    a = load_a_frag(A0, lda, mrow,      k0, koff);
    c00 = __builtin_amdgcn_wmma_f32_16x16x32_bf16(false, a, false, b, (short)0, c00, false, false);
    a = load_a_frag(A0, lda, mrow + 16, k0, koff);
    c01 = __builtin_amdgcn_wmma_f32_16x16x32_bf16(false, a, false, b, (short)0, c01, false, false);
    a = load_a_frag(A1, lda, mrow,      k0, koff);
    c10 = __builtin_amdgcn_wmma_f32_16x16x32_bf16(false, a, false, b, (short)0, c10, false, false);
    a = load_a_frag(A1, lda, mrow + 16, k0, koff);
    c11 = __builtin_amdgcn_wmma_f32_16x16x32_bf16(false, a, false, b, (short)0, c11, false, false);
  }
}

// ---- 2 stacked tiles (both M-halves, one subgraph) — LDS-B attention GEMMs -
__device__ __forceinline__ void wmma_tile2_bf16(const __bf16* A, int lda,
                                                const __bf16* B, int ldb,
                                                int n0, int Kdim, int lane,
                                                v8f& acc0, v8f& acc1) {
  const int mrow = lane & 15;
  const int koff = (lane >> 4) << 3;
  for (int k0 = 0; k0 < Kdim; k0 += 32) {
    const __bf16* pb = B + (long)(k0 + koff) * ldb + n0 + mrow;
    v16bf b;
    for (int i = 0; i < 8; ++i) { b[i] = pb[(long)i * ldb]; b[i + 8] = pb[(long)(i + 16) * ldb]; }
    v16bf a = load_a_frag(A, lda, mrow, k0, koff);
    acc0 = __builtin_amdgcn_wmma_f32_16x16x32_bf16(false, a, false, b, (short)0, acc0, false, false);
    a = load_a_frag(A, lda, mrow + 16, k0, koff);
    acc1 = __builtin_amdgcn_wmma_f32_16x16x32_bf16(false, a, false, b, (short)0, acc1, false, false);
  }
}

// ---- LayerNorm over H=256 for 32 tokens; 8 threads per token ---------------
template <typename OutT>
__device__ __forceinline__ void layer_norm(const float* xin, OutT* dst,
                                           const float* g, const float* b,
                                           int tid) {
  const int tok = tid >> 3;
  const int sub = tid & 7;
  const float* row = xin + tok * H_DIM;
  float s = 0.f, s2 = 0.f;
  for (int i = sub; i < H_DIM; i += 8) {
    float v = row[i];
    s += v; s2 += v * v;
  }
  s  += __shfl_xor(s,  1, 8); s  += __shfl_xor(s,  2, 8); s  += __shfl_xor(s,  4, 8);
  s2 += __shfl_xor(s2, 1, 8); s2 += __shfl_xor(s2, 2, 8); s2 += __shfl_xor(s2, 4, 8);
  const float mu   = s * (1.0f / H_DIM);
  const float var  = s2 * (1.0f / H_DIM) - mu * mu;
  const float rstd = rsqrtf(var + 1e-5f);
  for (int i = sub; i < H_DIM; i += 8) {
    float v = (row[i] - mu) * rstd * g[i] + b[i];
    dst[tok * H_DIM + i] = (OutT)v;
  }
}

// ---- weight fp32[N][K] -> bf16[K][N] transpose-convert ---------------------
__global__ void wcvt_kernel(const float* __restrict__ src,
                            __bf16* __restrict__ dst,
                            int Lc, int N, int K) {
  long idx = (long)blockIdx.x * blockDim.x + threadIdx.x;
  long total = (long)Lc * N * K;
  if (idx >= total) return;
  int l = (int)(idx / ((long)N * K));
  int rem = (int)(idx % ((long)N * K));
  int n = rem / K;
  int k = rem % K;
  dst[(long)l * N * K + (long)k * N + n] = (__bf16)src[idx];
}

// ---------------------------------------------------------------------------
__global__ __launch_bounds__(256)
void LocalSubgraphTransformer_kernel(
    const float* __restrict__ h, const unsigned char* __restrict__ valid,
    const int* __restrict__ eidx, const float* __restrict__ ea,
    const float* __restrict__ edge_W, const float* __restrict__ edge_b,
    const float* __restrict__ ln1_g, const float* __restrict__ ln1_b,
    const float* __restrict__ qkv_b, const float* __restrict__ out_b,
    const float* __restrict__ ln2_g, const float* __restrict__ ln2_b,
    const float* __restrict__ ff1_b, const float* __restrict__ ff2_b,
    const float* __restrict__ fnorm_g, const float* __restrict__ fnorm_b,
    const __bf16* __restrict__ qkvWt, const __bf16* __restrict__ outWt,
    const __bf16* __restrict__ ff1Wt, const __bf16* __restrict__ ff2Wt,
    float* __restrict__ out) {
  extern __shared__ char smem[];
  float*  xbufS[SG_PER_BLK];
  __bf16* biasS[SG_PER_BLK];
  __bf16* xnS[SG_PER_BLK];     // xn | obf timeshare
  __bf16* qS[SG_PER_BLK];
  __bf16* kS[SG_PER_BLK];
  __bf16* vS[SG_PER_BLK];
  char*   scrS[SG_PER_BLK];
#pragma unroll
  for (int g = 0; g < SG_PER_BLK; ++g) {
    char* base = smem + g * SG_STRIDE;
    xbufS[g] = (float*)(base + O_X);
    biasS[g] = (__bf16*)(base + O_BIAS);
    xnS[g]   = (__bf16*)(base + O_XN);
    qS[g]    = (__bf16*)(base + O_Q);
    kS[g]    = (__bf16*)(base + O_KT);
    vS[g]    = (__bf16*)(base + O_V);
    scrS[g]  = base + O_SCR;
  }
  float* negF = (float*)(smem + OFF_NEG);

  const int s0   = blockIdx.x * SG_PER_BLK;
  const int tid  = threadIdx.x;
  const int lane = tid & 31;
  const int wid  = tid >> 5;
  const int col  = lane & 15;
  const int mhi  = (lane >> 4) << 3;   // +8 rows for upper half-lanes in C/D

  // ---- load residual x (fp32), zero f32 bias accum, padding mask ----------
#pragma unroll
  for (int g = 0; g < SG_PER_BLK; ++g) {
    const float4* hp = (const float4*)(h + (size_t)(s0 + g) * K_TOK * H_DIM);
    float4* x4 = (float4*)xbufS[g];
    float4* a4 = (float4*)scrS[g];          // f32 bias accumulator
    const float4 z4 = {0.f, 0.f, 0.f, 0.f};
    for (int i = tid; i < (K_TOK * H_DIM) / 4; i += 256) {
      x4[i] = hp[i];
      a4[i] = z4;
    }
  }
  if (tid < SG_PER_BLK * K_TOK) {
    const int g = tid >> 5, t = tid & 31;
    negF[tid] = valid[(size_t)(s0 + g) * K_TOK + t] ? 0.0f : -1.0e30f;
  }
  __syncthreads();

  // ---- edge bias: proj [64,16]x[16,8] + ds_add_f32 scatter, then ->bf16 ---
  if (tid < SG_PER_BLK * EPS_E) {
    const int g = tid >> 6;
    const int e = (s0 + g) * EPS_E + (tid & 63);
    const int sv = eidx[e];
    const int dv = eidx[E_TOT + e];
    float* accF = (float*)scrS[g];
    float ear[16];
    for (int j = 0; j < 16; ++j) ear[j] = ea[(size_t)e * 16 + j];
    for (int hh = 0; hh < N_HEAD; ++hh) {
      float acc = edge_b[hh];
      for (int j = 0; j < 16; ++j) acc += ear[j] * edge_W[hh * 16 + j];
      atomicAdd(&accF[(hh * K_TOK + sv) * K_TOK + dv], acc);   // ds_add_f32
    }
  }
  __syncthreads();
#pragma unroll
  for (int g = 0; g < SG_PER_BLK; ++g) {
    const float* accF = (const float*)scrS[g];
    for (int i = tid; i < N_HEAD * K_TOK * K_TOK; i += 256)
      biasS[g][i] = (__bf16)accF[i];
  }
  __syncthreads();

  const float scale = 0.17677669529663687f;  // 1/sqrt(32)

  for (int l = 0; l < N_LAYER; ++l) {
    const float*  l1g = ln1_g + l * H_DIM;
    const float*  l1b = ln1_b + l * H_DIM;
    const float*  l2g = ln2_g + l * H_DIM;
    const float*  l2b = ln2_b + l * H_DIM;
    const float*  qkb = qkv_b + l * 3 * H_DIM;
    const float*  ob  = out_b + l * H_DIM;
    const float*  f1b = ff1_b + l * FFN_DIM;
    const float*  f2b = ff2_b + l * H_DIM;
    const __bf16* qW  = qkvWt + (size_t)l * H_DIM * 3 * H_DIM;
    const __bf16* oW  = outWt + (size_t)l * H_DIM * H_DIM;
    const __bf16* f1W = ff1Wt + (size_t)l * H_DIM * FFN_DIM;
    const __bf16* f2W = ff2Wt + (size_t)l * FFN_DIM * H_DIM;

    // ---- LN1 -> xn (bf16) --------------------------------------------------
#pragma unroll
    for (int g = 0; g < SG_PER_BLK; ++g)
      layer_norm<__bf16>(xbufS[g], xnS[g], l1g, l1b, tid);
    __syncthreads();

    // ---- QKV: [32,256]x2 @ [256,768]; scatter into q / kT / v --------------
    for (int nj = wid; nj < 48; nj += 8) {
      const int n0 = nj * 16;
      v8f c[SG_PER_BLK][2] = {};
      wmma_tile4_bf16(xnS[0], xnS[1], H_DIM, qW, 3 * H_DIM, n0, H_DIM, lane,
                      c[0][0], c[0][1], c[1][0], c[1][1]);
      const int which = n0 >> 8;          // uniform -> scalar branches
      const int hh    = (n0 & 255) >> 5;
      const int d     = (n0 & 31) + col;
      const float bv  = qkb[n0 + col];
#pragma unroll
      for (int g = 0; g < SG_PER_BLK; ++g)
#pragma unroll
        for (int half = 0; half < 2; ++half) {
          const int mb = half * 16 + mhi;
          for (int r = 0; r < 8; ++r) {
            const int m = mb + r;
            const __bf16 bvv = (__bf16)(c[g][half][r] + bv);
            if (which == 0)      qS[g][(hh * K_TOK + m) * D_HEAD + d] = bvv;
            else if (which == 1) kS[g][(hh * K_TOK + d) * K_TOK + m]  = bvv;
            else                 vS[g][(hh * K_TOK + m) * D_HEAD + d] = bvv;
          }
        }
    }
    __syncthreads();

    // ---- scores: wave -> (sg,head) pairs; q[32,32] @ kT -> bf16 scores -----
    for (int p = wid; p < SG_PER_BLK * N_HEAD; p += 8) {
      const int g  = p >> 3;
      const int hh = p & 7;
      const __bf16* qh = qS[g] + hh * K_TOK * D_HEAD;
      const __bf16* kh = kS[g] + hh * D_HEAD * K_TOK;
      __bf16* scoresB = (__bf16*)scrS[g];
      for (int nj = 0; nj < 2; ++nj) {
        const int n0 = nj * 16;
        v8f c0 = {}, c1 = {};
        wmma_tile2_bf16(qh, D_HEAD, kh, K_TOK, n0, D_HEAD, lane, c0, c1);
#pragma unroll
        for (int half = 0; half < 2; ++half) {
          const v8f acc = half ? c1 : c0;
          const int mb  = half * 16 + mhi;
          for (int r = 0; r < 8; ++r)
            scoresB[(hh * K_TOK + mb + r) * K_TOK + n0 + col] = (__bf16)acc[r];
        }
      }
    }
    __syncthreads();

    // ---- softmax: thread per (sg,head,row); bias + key padding -------------
    for (int rr = tid; rr < SG_PER_BLK * N_HEAD * K_TOK; rr += 256) {
      const int g  = rr >> 8;
      const int hh = (rr >> 5) & 7;
      const int m  = rr & 31;
      const __bf16* srow = (const __bf16*)scrS[g] + (hh * K_TOK + m) * K_TOK;
      const __bf16* brow = biasS[g] + (hh * K_TOK + m) * K_TOK;
      __bf16* arow = (__bf16*)(scrS[g] + 16384) + (hh * K_TOK + m) * K_TOK;
      const float* ng = negF + g * K_TOK;
      float mx = -3.0e38f;
      for (int j = 0; j < K_TOK; ++j)
        mx = fmaxf(mx, (float)srow[j] * scale + (float)brow[j] + ng[j]);
      float sum = 0.f;
      for (int j = 0; j < K_TOK; ++j)
        sum += __expf((float)srow[j] * scale + (float)brow[j] + ng[j] - mx);
      const float inv = 1.0f / sum;
      for (int j = 0; j < K_TOK; ++j) {
        float e = __expf((float)srow[j] * scale + (float)brow[j] + ng[j] - mx) * inv;
        arow[j] = (__bf16)e;
      }
    }
    __syncthreads();

    // ---- O = attn @ v, concat heads into obf (= xn region) -----------------
    for (int p = wid; p < SG_PER_BLK * N_HEAD; p += 8) {
      const int g  = p >> 3;
      const int hh = p & 7;
      const __bf16* ah = (const __bf16*)(scrS[g] + 16384) + hh * K_TOK * K_TOK;
      const __bf16* vh = vS[g] + hh * K_TOK * D_HEAD;
      __bf16* obf = xnS[g];
      for (int nj = 0; nj < 2; ++nj) {
        const int n0 = nj * 16;
        v8f c0 = {}, c1 = {};
        wmma_tile2_bf16(ah, K_TOK, vh, D_HEAD, n0, K_TOK, lane, c0, c1);
#pragma unroll
        for (int half = 0; half < 2; ++half) {
          const v8f acc = half ? c1 : c0;
          const int mb  = half * 16 + mhi;
          for (int r = 0; r < 8; ++r)
            obf[(mb + r) * H_DIM + hh * D_HEAD + n0 + col] = (__bf16)acc[r];
        }
      }
    }
    __syncthreads();

    // ---- out-proj + residual: x += obf @ outWt + out_b ---------------------
    for (int nj = wid; nj < 16; nj += 8) {
      const int n0 = nj * 16;
      v8f c[SG_PER_BLK][2] = {};
      wmma_tile4_bf16(xnS[0], xnS[1], H_DIM, oW, H_DIM, n0, H_DIM, lane,
                      c[0][0], c[0][1], c[1][0], c[1][1]);
      const float bv = ob[n0 + col];
#pragma unroll
      for (int g = 0; g < SG_PER_BLK; ++g)
#pragma unroll
        for (int half = 0; half < 2; ++half) {
          const int mb = half * 16 + mhi;
          for (int r = 0; r < 8; ++r)
            xbufS[g][(mb + r) * H_DIM + n0 + col] += c[g][half][r] + bv;
        }
    }
    __syncthreads();

    // ---- LN2 -> xn (bf16) --------------------------------------------------
#pragma unroll
    for (int g = 0; g < SG_PER_BLK; ++g)
      layer_norm<__bf16>(xbufS[g], xnS[g], l2g, l2b, tid);
    __syncthreads();

    // ---- FFN1 + relu -> hid bf16 [32,512] (scratch) ------------------------
    for (int nj = wid; nj < 32; nj += 8) {
      const int n0 = nj * 16;
      v8f c[SG_PER_BLK][2] = {};
      wmma_tile4_bf16(xnS[0], xnS[1], H_DIM, f1W, FFN_DIM, n0, H_DIM, lane,
                      c[0][0], c[0][1], c[1][0], c[1][1]);
      const float bv = f1b[n0 + col];
#pragma unroll
      for (int g = 0; g < SG_PER_BLK; ++g) {
        __bf16* hid = (__bf16*)scrS[g];
#pragma unroll
        for (int half = 0; half < 2; ++half) {
          const int mb = half * 16 + mhi;
          for (int r = 0; r < 8; ++r)
            hid[(mb + r) * FFN_DIM + n0 + col] =
                (__bf16)fmaxf(c[g][half][r] + bv, 0.0f);
        }
      }
    }
    __syncthreads();

    // ---- FFN2 + residual: x += hid @ ff2Wt + ff2_b -------------------------
    for (int nj = wid; nj < 16; nj += 8) {
      const int n0 = nj * 16;
      v8f c[SG_PER_BLK][2] = {};
      wmma_tile4_bf16((const __bf16*)scrS[0], (const __bf16*)scrS[1], FFN_DIM,
                      f2W, H_DIM, n0, FFN_DIM, lane,
                      c[0][0], c[0][1], c[1][0], c[1][1]);
      const float bv = f2b[n0 + col];
#pragma unroll
      for (int g = 0; g < SG_PER_BLK; ++g)
#pragma unroll
        for (int half = 0; half < 2; ++half) {
          const int mb = half * 16 + mhi;
          for (int r = 0; r < 8; ++r)
            xbufS[g][(mb + r) * H_DIM + n0 + col] += c[g][half][r] + bv;
        }
    }
    __syncthreads();
  }

  // ---- final LN (fp32, into scratch) + masked mean pool --------------------
#pragma unroll
  for (int g = 0; g < SG_PER_BLK; ++g)
    layer_norm<float>(xbufS[g], (float*)scrS[g], fnorm_g, fnorm_b, tid);
  __syncthreads();
  for (int rr = tid; rr < SG_PER_BLK * H_DIM; rr += 256) {
    const int g = rr >> 8;
    const int c = rr & (H_DIM - 1);
    const float* fn = (const float*)scrS[g];
    const float* ng = negF + g * K_TOK;
    float sum = 0.f, cnt = 0.f;
    for (int m = 0; m < K_TOK; ++m) {
      if (ng[m] == 0.0f) {
        sum += fn[m * H_DIM + c];
        cnt += 1.0f;
      }
    }
    out[(size_t)(s0 + g) * H_DIM + c] = sum / fmaxf(cnt, 1.0f);
  }
}

// ---------------------------------------------------------------------------
extern "C" void kernel_launch(void* const* d_in, const int* in_sizes, int n_in,
                              void* d_out, int out_size, void* d_ws, size_t ws_size,
                              hipStream_t stream) {
  const float* h       = (const float*)d_in[0];
  const unsigned char* valid = (const unsigned char*)d_in[1];
  const int*   eidx    = (const int*)d_in[2];
  const float* ea      = (const float*)d_in[3];
  // d_in[4]=edge_ptr, d_in[5]=S, d_in[6]=k : compile-time constants here
  const float* edge_W  = (const float*)d_in[7];
  const float* edge_b  = (const float*)d_in[8];
  const float* ln1_g   = (const float*)d_in[9];
  const float* ln1_b   = (const float*)d_in[10];
  const float* qkv_W   = (const float*)d_in[11];
  const float* qkv_b   = (const float*)d_in[12];
  const float* out_W   = (const float*)d_in[13];
  const float* out_b   = (const float*)d_in[14];
  const float* ln2_g   = (const float*)d_in[15];
  const float* ln2_b   = (const float*)d_in[16];
  const float* ff1_W   = (const float*)d_in[17];
  const float* ff1_b   = (const float*)d_in[18];
  const float* ff2_W   = (const float*)d_in[19];
  const float* ff2_b   = (const float*)d_in[20];
  const float* fnorm_g = (const float*)d_in[21];
  const float* fnorm_b = (const float*)d_in[22];
  float* out = (float*)d_out;

  // Workspace: bf16 transposed weights [K][N] per layer.
  __bf16* qkvWt = (__bf16*)d_ws;                                  // 4*256*768
  __bf16* outWt = qkvWt + (size_t)N_LAYER * H_DIM * 3 * H_DIM;    // 4*256*256
  __bf16* ff1Wt = outWt + (size_t)N_LAYER * H_DIM * H_DIM;        // 4*256*512
  __bf16* ff2Wt = ff1Wt + (size_t)N_LAYER * H_DIM * FFN_DIM;      // 4*512*256

  const int B = 256;
  wcvt_kernel<<<(N_LAYER * 3 * H_DIM * H_DIM + B - 1) / B, B, 0, stream>>>(
      qkv_W, qkvWt, N_LAYER, 3 * H_DIM, H_DIM);
  wcvt_kernel<<<(N_LAYER * H_DIM * H_DIM + B - 1) / B, B, 0, stream>>>(
      out_W, outWt, N_LAYER, H_DIM, H_DIM);
  wcvt_kernel<<<(N_LAYER * FFN_DIM * H_DIM + B - 1) / B, B, 0, stream>>>(
      ff1_W, ff1Wt, N_LAYER, FFN_DIM, H_DIM);
  wcvt_kernel<<<(N_LAYER * H_DIM * FFN_DIM + B - 1) / B, B, 0, stream>>>(
      ff2_W, ff2Wt, N_LAYER, H_DIM, FFN_DIM);

  hipFuncSetAttribute((const void*)LocalSubgraphTransformer_kernel,
                      hipFuncAttributeMaxDynamicSharedMemorySize, SMEM_BYTES);

  LocalSubgraphTransformer_kernel<<<S_TOT / SG_PER_BLK, 256, SMEM_BYTES, stream>>>(
      h, valid, eidx, ea, edge_W, edge_b, ln1_g, ln1_b, qkv_b, out_b,
      ln2_g, ln2_b, ff1_b, ff2_b, fnorm_g, fnorm_b,
      qkvWt, outWt, ff1Wt, ff2Wt, out);
}
